// BFConv2d_72035191488550
// MI455X (gfx1250) — matrix operand order repaired and checked by reading, not compiled, not run on hardware
//
#include <hip/hip_runtime.h>

// ---------------------------------------------------------------------------
// BFP-quantized 3x3 conv for MI455X (gfx1250), wave32 + WMMA.
//   x: [16,32,224,224] f32   weight: [32,32,3,3] f32   bias: [32] f32
//   out = bfp8( conv(bfp8(x), bfp8(w)) + bias )   (blocks of 36 flat elems)
// BFP-8 values are m * 2^(e-7), |m| <= 256  ->  exactly representable in f16,
// so v_wmma_f32_16x16x32_f16 (f32 accumulate) reproduces the f32 reference.
// Conv kernel: implicit GEMM, M=32 (Cout) x N=32 (pixels) x 4 output rows per
// 128-thread block; K = 9 taps x 32 Cin. Weights + input halo staged in LDS,
// each A fragment reused across the 4 row-accumulators (36 WMMAs / wave).
// ---------------------------------------------------------------------------

typedef _Float16 v8h  __attribute__((ext_vector_type(8)));
typedef _Float16 v16h __attribute__((ext_vector_type(16)));
typedef float    v8f  __attribute__((ext_vector_type(8)));

#define CIN   32
#define COUT  32
#define IMH   224
#define IMW   224
#define NIMG  16
#define NTW   32            // output pixels (w) per block
#define ROWS  4             // output rows (h) per block (224 = 4*56)
#define XROWS (ROWS + 2)    // input rows incl. halo
#define XCOLS 34            // NTW + 2 halo columns
#define CIPAD 40            // padded ci stride (16B-aligned, spreads banks)

__device__ __forceinline__ float wave_max32(float m) {
#pragma unroll
  for (int off = 16; off > 0; off >>= 1)
    m = fmaxf(m, __shfl_xor(m, off, 32));
  return m;
}

// ---- BFP quantize: one wave32 per group of 36 flat elements ---------------
template <typename OutT>
__global__ __launch_bounds__(256) void bfp_quant_kernel(
    const float* __restrict__ in, OutT* __restrict__ out, long n, long ngroups) {
  long g = (long)blockIdx.x * (blockDim.x >> 5) + (threadIdx.x >> 5);
  if (g >= ngroups) return;
  int  lane = threadIdx.x & 31;
  long base = g * 36;
  long i0 = base + lane;
  long i1 = base + 32 + lane;            // lanes 0..3 cover elements 32..35
  float v0 = (i0 < n) ? in[i0] : 0.0f;
  float v1 = (lane < 4 && i1 < n) ? in[i1] : 0.0f;
  float mx = wave_max32(fmaxf(fabsf(v0), fabsf(v1)));
  if (mx > 0.0f) {
    int ex;
    (void)frexpf(mx, &ex);               // mx = f*2^ex, f in [0.5,1): floor(log2)=ex-1
    int sh = 8 - ex;                     // 1/scale = 2^(7-(ex-1)) = 2^sh
    v0 = ldexpf(rintf(ldexpf(v0, sh)), -sh);   // exact pow2 scaling, RNE round
    v1 = ldexpf(rintf(ldexpf(v1, sh)), -sh);
  } else {
    v0 = 0.0f; v1 = 0.0f;
  }
  if (i0 < n) out[i0] = (OutT)v0;
  if (lane < 4 && i1 < n) out[i1] = (OutT)v1;
}

// ---- Weight quantize: 9216 elems = 256 groups; remap OIHW -> [tap][co][ci]
__global__ __launch_bounds__(256) void bfp_quant_weight_kernel(
    const float* __restrict__ in, _Float16* __restrict__ out) {
  int g = blockIdx.x * (blockDim.x >> 5) + (threadIdx.x >> 5);
  if (g >= 256) return;
  int lane = threadIdx.x & 31;
  int i0 = g * 36 + lane;
  int i1 = g * 36 + 32 + lane;
  float v0 = in[i0];
  float v1 = (lane < 4) ? in[i1] : 0.0f;
  float mx = wave_max32(fmaxf(fabsf(v0), fabsf(v1)));
  if (mx > 0.0f) {
    int ex;
    (void)frexpf(mx, &ex);
    int sh = 8 - ex;
    v0 = ldexpf(rintf(ldexpf(v0, sh)), -sh);
    v1 = ldexpf(rintf(ldexpf(v1, sh)), -sh);
  } else {
    v0 = 0.0f; v1 = 0.0f;
  }
  // OIHW flat f = ((co*32+ci)*3+kh)*3+kw  ->  [(kh*3+kw)][co][ci]
  {
    int co = i0 / 288, rem = i0 % 288, ci = rem / 9, t = rem % 9;
    out[(t * COUT + co) * CIN + ci] = (_Float16)v0;
  }
  if (lane < 4) {
    int co = i1 / 288, rem = i1 % 288, ci = rem / 9, t = rem % 9;
    out[(t * COUT + co) * CIN + ci] = (_Float16)v1;
  }
}

// ---- Implicit-GEMM conv: 32co x 32pix x 4 rows per block ------------------
__global__ __launch_bounds__(128) void bfp_conv_wmma_kernel(
    const _Float16* __restrict__ xq,   // [n][ci][h][w] f16 (quantized)
    const _Float16* __restrict__ wq,   // [tap][co][ci] f16 (quantized)
    const float*    __restrict__ bias, // [co]
    float*          __restrict__ y) {  // [n][co][h][w] f32 (pre-out-quant)
  __shared__ __attribute__((aligned(16))) _Float16 w_lds[9][COUT][CIN];          // 18 KB
  __shared__ __attribute__((aligned(16))) _Float16 x_lds[XROWS][XCOLS][CIPAD];   // 16 KB

  int b  = blockIdx.x;
  int wc = b % 7;                   // 224 / 32
  int hb = (b / 7) % (IMH / ROWS);  // row-block index
  int n  = b / (7 * (IMH / ROWS));
  int h0 = hb * ROWS;
  int wbase = wc * NTW;
  int tid = threadIdx.x;

  // Stage all quantized weights (coalesced dword copy: 4608 u32).
  {
    const uint32_t* src = (const uint32_t*)wq;
    uint32_t*       dst = (uint32_t*)&w_lds[0][0][0];
    for (int i = tid; i < 9 * COUT * CIN / 2; i += 128) dst[i] = src[i];
  }
  // Stage the (ROWS+2)-row input halo tile, ci innermost, zero-padded borders.
  for (int i = tid; i < XROWS * CIN * XCOLS; i += 128) {
    int c   = i % XCOLS;
    int ci  = (i / XCOLS) % CIN;
    int r   = i / (XCOLS * CIN);
    int hin = h0 + r - 1;
    int win = wbase + c - 1;
    _Float16 v = (_Float16)0.0f;
    if (hin >= 0 && hin < IMH && win >= 0 && win < IMW)
      v = xq[(((long)n * CIN + ci) * IMH + hin) * IMW + win];
    x_lds[r][c][ci] = v;
  }
  __syncthreads();

  int waveId = tid >> 5;
  int lane   = tid & 31;
  int mtile  = waveId & 1;          // co half  (0: co 0..15, 1: co 16..31)
  int ntile  = waveId >> 1;         // pixel half
  int hi     = lane >> 4;           // lane half per ISA fragment tables
  int l15    = lane & 15;

  // A(16x32 f16): lane holds M=l15; elems 0..7 -> K = hi*8+0..7, 8..15 -> +16.
  // B(32x16 f16): lane holds N=l15; elems 0..7 -> K = hi*8+0..7, 8..15 -> +16.
  int coA = mtile * 16 + l15;
  int ciA = hi * 8;
  int ciB = hi * 8;
  int xc  = ntile * 16 + l15;       // base halo column for this lane

  v8f acc[ROWS];
#pragma unroll
  for (int rr = 0; rr < ROWS; ++rr) acc[rr] = (v8f){};

#pragma unroll
  for (int t = 0; t < 9; ++t) {
    int kh = t / 3, kw = t % 3;
    v8h a_lo = *(const v8h*)&w_lds[t][coA][ciA];          // ds_load_b128
    v8h a_hi = *(const v8h*)&w_lds[t][coA][ciA + 16];
    v16h a = __builtin_shufflevector(a_lo, a_hi, 0, 1, 2, 3, 4, 5, 6, 7,
                                     8, 9, 10, 11, 12, 13, 14, 15);
#pragma unroll
    for (int rr = 0; rr < ROWS; ++rr) {                   // A reused 4x
      v8h b_lo = *(const v8h*)&x_lds[rr + kh][xc + kw][ciB];
      v8h b_hi = *(const v8h*)&x_lds[rr + kh][xc + kw][ciB + 16];
      v16h bm = __builtin_shufflevector(b_lo, b_hi, 0, 1, 2, 3, 4, 5, 6, 7,
                                        8, 9, 10, 11, 12, 13, 14, 15);
      acc[rr] = __builtin_amdgcn_wmma_f32_16x16x32_f16(
          /*neg_a=*/false, a, /*neg_b=*/false, bm,
          /*c_mod=*/(short)0, acc[rr], /*reuse_a=*/false, /*reuse_b=*/false);
    }
  }

  // C/D layout: N = l15, M = r + 8*hi (per VGPR r). Add bias, store f32.
  int np = wbase + ntile * 16 + l15;
  float bia[8];
#pragma unroll
  for (int r = 0; r < 8; ++r) bia[r] = bias[mtile * 16 + hi * 8 + r];
#pragma unroll
  for (int rr = 0; rr < ROWS; ++rr) {
#pragma unroll
    for (int r = 0; r < 8; ++r) {
      int m = mtile * 16 + hi * 8 + r;
      y[(((long)n * COUT + m) * IMH + (h0 + rr)) * IMW + np] = acc[rr][r] + bia[r];
    }
  }
}

// ---------------------------------------------------------------------------
extern "C" void kernel_launch(void* const* d_in, const int* in_sizes, int n_in,
                              void* d_out, int out_size, void* d_ws, size_t ws_size,
                              hipStream_t stream) {
  const float* x    = (const float*)d_in[0];
  const float* w    = (const float*)d_in[1];
  const float* bias = (const float*)d_in[2];
  float*       out  = (float*)d_out;

  const long NX = (long)NIMG * CIN * IMH * IMW;      // 25,690,112 (== out_size)

  // Workspace: [ y f32 | xq f16 | wq f16 ]
  float*    y  = (float*)d_ws;
  _Float16* xq = (_Float16*)((char*)d_ws + NX * sizeof(float));
  _Float16* wq = (_Float16*)((char*)d_ws + NX * sizeof(float) + NX * sizeof(_Float16));

  long ngroups = (NX + 35) / 36;                     // 713,615 (zero-padded tail)
  int  gpb     = 256 / 32;                           // groups per 8-wave block
  int  qblocks = (int)((ngroups + gpb - 1) / gpb);

  bfp_quant_kernel<_Float16><<<qblocks, 256, 0, stream>>>(x, xq, NX, ngroups);
  bfp_quant_weight_kernel<<<32, 256, 0, stream>>>(w, wq);
  bfp_conv_wmma_kernel<<<7 * (IMH / ROWS) * NIMG, 128, 0, stream>>>(xq, wq, bias, y);
  bfp_quant_kernel<float><<<qblocks, 256, 0, stream>>>(y, out, NX, ngroups);
}
